// AVAlign_31379031065120
// MI455X (gfx1250) — matrix operand order repaired and checked by reading, not compiled, not run on hardware
//
#include <hip/hip_runtime.h>

// ---------------------------------------------------------------------------
// Problem constants (from reference)
// ---------------------------------------------------------------------------
#define S_   32
#define FR_  8
#define C_   28
#define HW_  14
#define NHW_ 196          // HW*HW
#define NHWP_ 224         // padded to multiple of 32 for WMMA K loop
#define CIN_ 1024
#define CMID_ 512
#define D_   128
#define T_   16
#define B_A  (S_*C_)      // 896
#define B_V  (S_*FR_)     // 256
#define ROWS_TV (B_V*C_)  // 7168
#define EPS_ 1e-10f
#define LPAD 40           // LDS row stride (ushorts): 80B = 5*16B, 20-bank step

typedef __attribute__((ext_vector_type(16))) __bf16  bf16x16;
typedef __attribute__((ext_vector_type(8)))  float   v8f;
typedef __attribute__((ext_vector_type(4)))  unsigned int u32x4;
typedef __attribute__((ext_vector_type(4)))  float   f32x4;
typedef __attribute__((ext_vector_type(4)))  int     i32x4;

union BFrag { bf16x16 v; unsigned short u[16]; u32x4 q[2]; };

__device__ __forceinline__ unsigned short f2bf(float f) {
    union { __bf16 b; unsigned short u; } c;
    c.b = (__bf16)f;                    // native f32->bf16 convert
    return c.u;
}

__device__ __forceinline__ v8f wmma_bf16(const BFrag& a, const BFrag& b, v8f c) {
    return __builtin_amdgcn_wmma_f32_16x16x32_bf16(
        false, a.v, false, b.v, (short)0, c, false, false);
}

// ---- CDNA5 async copy: 16B per lane, global -> LDS, ASYNCcnt-tracked -------
__device__ __forceinline__ void async_cp16(const void* g, void* l) {
#if defined(__HIP_DEVICE_COMPILE__) && __has_builtin(__builtin_amdgcn_global_load_async_to_lds_b128)
    typedef __attribute__((address_space(1))) i32x4 gv4i;
    typedef __attribute__((address_space(3))) i32x4 lv4i;
    __builtin_amdgcn_global_load_async_to_lds_b128(
        (gv4i*)const_cast<void*>(g), (lv4i*)l, 0, 0);
#else
    unsigned lofs = (unsigned)(unsigned long long)
        (__attribute__((address_space(3))) void*)l;
    asm volatile("global_load_async_to_lds_b128 %0, %1, off"
                 :: "v"(lofs), "v"((unsigned long long)g) : "memory");
#endif
}

__device__ __forceinline__ void wait_async0() {
#if defined(__HIP_DEVICE_COMPILE__) && __has_builtin(__builtin_amdgcn_s_wait_asynccnt)
    __builtin_amdgcn_s_wait_asynccnt(0);
#else
    asm volatile("s_wait_asynccnt 0x0" ::: "memory");
#endif
}

// ---------------------------------------------------------------------------
// Kernel 0: f32 -> bf16 conversion (weights, done once)
// ---------------------------------------------------------------------------
__global__ void k_cvt(const float* __restrict__ src,
                      unsigned short* __restrict__ dst, int n)
{
    int i = blockIdx.x * 256 + threadIdx.x;
    if (i < n) dst[i] = f2bf(src[i]);
}

// ---------------------------------------------------------------------------
// Kernel 1: a[b, o] = max_t( (Wt @ feat_a[b])[o, t] ) + bt[o]
// One block per batch b (896). 8 waves x 4 M-tiles = 512 rows. K=1024.
// Weight chunk staged via GLOBAL_LOAD_ASYNC_TO_LDS_B128 (bf16, pre-converted).
// ---------------------------------------------------------------------------
__global__ __launch_bounds__(256) void k_gemm_a(
    const float* __restrict__ featA, const unsigned short* __restrict__ WtBf,
    const float* __restrict__ bt, float* __restrict__ aOut)
{
    __shared__ unsigned short sW[CMID_ * LPAD];  // [o][k] padded, 40 KB
    __shared__ unsigned short sF[T_ * LPAD];     // [t][k] padded (transposed!)

    const int b    = blockIdx.x;
    const int tid  = threadIdx.x;
    const int lane = tid & 31;
    const int wave = tid >> 5;
    const float* fbase = featA + (size_t)b * CIN_ * T_;

    v8f acc[4] = {};

    const int mrow = lane & 15;
    const int kgrp = (lane >> 4) * 8;   // A: K-group per lane half
    const int ncol = lane & 15;
    const int khB  = (lane >> 4) * 8;   // B: K-group per lane half

    for (int ks = 0; ks < CIN_ / 32; ++ks) {
        const int kb = ks * 32;
        // Async-stage weight chunk: 512 rows x 32 bf16 (4 x 16B groups per row)
        #pragma unroll
        for (int it = 0; it < 8; ++it) {
            const int g = it * 256 + tid;
            const int o = g >> 2, sub = g & 3;
            async_cp16(WtBf + (size_t)o * CIN_ + kb + sub * 8,
                       &sW[o * LPAD + sub * 8]);
        }
        // Stage feat chunk with f32->bf16 convert, transposed to [t][k]
        for (int idx = tid; idx < 32 * T_; idx += 256) {
            const int k = idx >> 4, t = idx & 15;
            sF[t * LPAD + k] = f2bf(fbase[(size_t)kb * T_ + idx]);
        }
        wait_async0();
        __syncthreads();

        // B fragment: 8 ushorts at k=8h and 8 at k=16+8h -> two b128 LDS loads
        BFrag bfb;
        bfb.q[0] = *(const u32x4*)&sF[ncol * LPAD + khB];
        bfb.q[1] = *(const u32x4*)&sF[ncol * LPAD + khB + 16];

        #pragma unroll
        for (int t = 0; t < 4; ++t) {
            const int o = (wave * 4 + t) * 16 + mrow;
            BFrag af;
            af.q[0] = *(const u32x4*)&sW[o * LPAD + kgrp];
            af.q[1] = *(const u32x4*)&sW[o * LPAD + 16 + kgrp];
            acc[t] = wmma_bf16(af, bfb, acc[t]);
        }
        __syncthreads();
    }

    // Epilogue: max over N (t) within each 16-lane group, + bias, store
    #pragma unroll
    for (int t = 0; t < 4; ++t) {
        #pragma unroll
        for (int i = 0; i < 8; ++i) {
            float v = acc[t][i];
            #pragma unroll
            for (int off = 1; off < 16; off <<= 1)
                v = fmaxf(v, __shfl_xor(v, off, 32));
            const int o = (wave * 4 + t) * 16 + i + ((lane >> 4) << 3);
            if ((lane & 15) == 0)
                aOut[(size_t)b * CMID_ + o] = v + bt[o];
        }
    }
}

// ---------------------------------------------------------------------------
// Kernel 2: v[b, o, hw] = (Ws @ feat_v[b])[o, hw] + bs[o], stored bf16 as
// vT[b, o, hw] with hw padded to 224 (pad entries = 0) so k_fg's B-fragments
// are contiguous. Grid: (256 batches, 14 hw-tiles).
// ---------------------------------------------------------------------------
__global__ __launch_bounds__(256) void k_gemm_v(
    const float* __restrict__ featV, const unsigned short* __restrict__ WsBf,
    const float* __restrict__ bs, unsigned short* __restrict__ vT)
{
    __shared__ unsigned short sW[CMID_ * LPAD];
    __shared__ unsigned short sF[16 * LPAD];

    const int b  = blockIdx.x;
    const int nt = blockIdx.y;          // hw tile 0..13
    const int tid = threadIdx.x;
    const int lane = tid & 31;
    const int wave = tid >> 5;
    const float* fbase = featV + (size_t)b * CIN_ * NHW_;

    v8f acc[4] = {};

    const int mrow = lane & 15;
    const int kgrp = (lane >> 4) * 8;
    const int ncol = lane & 15;
    const int khB  = (lane >> 4) * 8;

    for (int ks = 0; ks < CIN_ / 32; ++ks) {
        const int kb = ks * 32;
        #pragma unroll
        for (int it = 0; it < 8; ++it) {
            const int g = it * 256 + tid;
            const int o = g >> 2, sub = g & 3;
            async_cp16(WsBf + (size_t)o * CIN_ + kb + sub * 8,
                       &sW[o * LPAD + sub * 8]);
        }
        for (int idx = tid; idx < 32 * 16; idx += 256) {
            const int k = idx >> 4, cc = idx & 15;
            const int col = nt * 16 + cc;
            float val = (col < NHW_) ? fbase[(size_t)(kb + k) * NHW_ + col] : 0.0f;
            sF[cc * LPAD + k] = f2bf(val);   // transposed: [n][k]
        }
        wait_async0();
        __syncthreads();

        BFrag bfb;
        bfb.q[0] = *(const u32x4*)&sF[ncol * LPAD + khB];
        bfb.q[1] = *(const u32x4*)&sF[ncol * LPAD + khB + 16];

        #pragma unroll
        for (int t = 0; t < 4; ++t) {
            const int o = (wave * 4 + t) * 16 + mrow;
            BFrag af;
            af.q[0] = *(const u32x4*)&sW[o * LPAD + kgrp];
            af.q[1] = *(const u32x4*)&sW[o * LPAD + 16 + kgrp];
            acc[t] = wmma_bf16(af, bfb, acc[t]);
        }
        __syncthreads();
    }

    const int hw = nt * 16 + (lane & 15);
    #pragma unroll
    for (int t = 0; t < 4; ++t) {
        #pragma unroll
        for (int i = 0; i < 8; ++i) {
            const int o = (wave * 4 + t) * 16 + i + ((lane >> 4) << 3);
            unsigned short w = (hw < NHW_) ? f2bf(acc[t][i] + bs[o])
                                           : (unsigned short)0;
            vT[((size_t)b * CMID_ + o) * NHWP_ + hw] = w;
        }
    }
}

// ---------------------------------------------------------------------------
// Kernel 3: camsum[b, c] = sum_hw cam[b, c, hw]
// ---------------------------------------------------------------------------
__global__ void k_camsum(const float* __restrict__ cam, float* __restrict__ camsum)
{
    const int b = blockIdx.x;
    const int c = threadIdx.x;
    if (c < C_) {
        const float* p = cam + ((size_t)b * C_ + c) * NHW_;
        float s = 0.f;
        for (int i = 0; i < NHW_; ++i) s += p[i];
        camsum[b * C_ + c] = s;
    }
}

// ---------------------------------------------------------------------------
// Kernel 4: fg[b, c, o] = (cam2[b](28x196) @ v[b]^T(196x512))[c, o] / (sum+eps)
// M padded 28->32 (2 tiles), K padded 196->224 (7 steps), N=512 (8 waves x 4).
// B-fragments are contiguous b128 global loads from vT[b][o][hw].
// ---------------------------------------------------------------------------
__global__ __launch_bounds__(256) void k_fg(
    const float* __restrict__ cam, const unsigned short* __restrict__ vT,
    const float* __restrict__ camsum, float* __restrict__ fg)
{
    const int b = blockIdx.x;
    const int lane = threadIdx.x & 31;
    const int wave = threadIdx.x >> 5;

    v8f acc[2][4] = {};

    const int mrow = lane & 15;
    const int kgrp = (lane >> 4) * 8;
    const int ncolL = lane & 15;
    const int khB  = (lane >> 4) * 8;

    for (int ks = 0; ks < NHWP_ / 32; ++ks) {
        const int kb = ks * 32;
        // A fragments (2 M-tiles) from cam, f32->bf16, zero-padded
        BFrag af[2];
        #pragma unroll
        for (int mt = 0; mt < 2; ++mt) {
            const int c = mt * 16 + mrow;
            #pragma unroll
            for (int j = 0; j < 16; ++j) {
                const int k = kb + ((j < 8) ? (kgrp + j) : (16 + kgrp + (j - 8)));
                float x = (c < C_ && k < NHW_)
                          ? cam[((size_t)b * C_ + c) * NHW_ + k] : 0.0f;
                af[mt].u[j] = f2bf(x);
            }
        }
        #pragma unroll
        for (int ntl = 0; ntl < 4; ++ntl) {
            const int o = (wave * 4 + ntl) * 16 + ncolL;
            const unsigned short* pb =
                vT + ((size_t)b * CMID_ + o) * NHWP_ + kb + khB;
            BFrag bfb;
            bfb.q[0] = *(const u32x4*)pb;
            bfb.q[1] = *(const u32x4*)(pb + 16);
            #pragma unroll
            for (int mt = 0; mt < 2; ++mt)
                acc[mt][ntl] = wmma_bf16(af[mt], bfb, acc[mt][ntl]);
        }
    }

    #pragma unroll
    for (int mt = 0; mt < 2; ++mt)
        #pragma unroll
        for (int ntl = 0; ntl < 4; ++ntl)
            #pragma unroll
            for (int i = 0; i < 8; ++i) {
                const int c = mt * 16 + i + ((lane >> 4) << 3);
                const int o = (wave * 4 + ntl) * 16 + (lane & 15);
                if (c < C_) {
                    const float scale = 1.0f / (camsum[b * C_ + c] + EPS_);
                    fg[((size_t)b * C_ + c) * CMID_ + o] = acc[mt][ntl][i] * scale;
                }
            }
}

// ---------------------------------------------------------------------------
// Kernel 5: generic projection out(R x 128) = X(R x 512) @ W(128 x 512)^T + b
// One block per 16-row tile; 8 waves = 8 N-tiles. W pre-converted to bf16.
// ---------------------------------------------------------------------------
__global__ __launch_bounds__(256) void k_proj(
    const float* __restrict__ X, const unsigned short* __restrict__ WBf,
    const float* __restrict__ bias, float* __restrict__ out)
{
    const int mt = blockIdx.x;
    const int lane = threadIdx.x & 31;
    const int wave = threadIdx.x >> 5;   // N tile

    v8f acc = {};
    const int r = mt * 16 + (lane & 15);       // A row
    const int n = wave * 16 + (lane & 15);     // B col (d)
    const int kgrp = (lane >> 4) * 8;
    const int khB  = (lane >> 4) * 8;

    for (int ks = 0; ks < CMID_ / 32; ++ks) {
        const int kb = ks * 32;
        // A: 16 floats via four b128 loads, convert to bf16
        const f32x4* px = (const f32x4*)(X + (size_t)r * CMID_ + kb);
        const int q0 = kgrp >> 2;
        f32x4 x0 = px[q0], x1 = px[q0 + 1], x2 = px[q0 + 4], x3 = px[q0 + 5];
        BFrag af;
        #pragma unroll
        for (int j = 0; j < 4; ++j) {
            af.u[j]      = f2bf(x0[j]);
            af.u[4 + j]  = f2bf(x1[j]);
            af.u[8 + j]  = f2bf(x2[j]);
            af.u[12 + j] = f2bf(x3[j]);
        }
        // B: 8 bf16 at k=8h and 8 at k=16+8h -> two b128 loads
        const unsigned short* pw = WBf + (size_t)n * CMID_ + kb + khB;
        BFrag bfb;
        bfb.q[0] = *(const u32x4*)pw;
        bfb.q[1] = *(const u32x4*)(pw + 16);
        acc = wmma_bf16(af, bfb, acc);
    }

    #pragma unroll
    for (int i = 0; i < 8; ++i) {
        const int rr = mt * 16 + i + ((lane >> 4) << 3);
        const int d  = wave * 16 + (lane & 15);
        out[(size_t)rr * D_ + d] = acc[i] + bias[d];
    }
}

// ---------------------------------------------------------------------------
// Kernel 6: loss_co[s,f,c] = mean_d (ta[s,c,d]-tv[s,f,c,d])^2 * (label[s,c]>0)
// ---------------------------------------------------------------------------
__global__ void k_loss_co(const float* __restrict__ ta, const float* __restrict__ tv,
                          const int* __restrict__ label, float* __restrict__ out)
{
    const int idx = blockIdx.x * blockDim.x + threadIdx.x;
    if (idx >= S_ * FR_ * C_) return;
    const int c = idx % C_;
    const int s = idx / (C_ * FR_);
    const float* pa = ta + ((size_t)s * C_ + c) * D_;
    const float* pv = tv + (size_t)idx * D_;     // idx == (s*FR+f)*C + c
    float acc = 0.f;
    for (int d = 0; d < D_; ++d) {
        const float df = pa[d] - pv[d];
        acc += df * df;
    }
    const float mask = (label[s * C_ + c] > 0) ? 1.0f : 0.0f;
    out[idx] = acc * (1.0f / D_) * mask;
}

// ---------------------------------------------------------------------------
// Kernel 7: loss_di[s,i,f,j] = na2 + nv2 - 2*dot/D, fused single pass, masked.
// ---------------------------------------------------------------------------
__global__ void k_loss_di(const float* __restrict__ ta, const float* __restrict__ tv,
                          const int* __restrict__ label, float* __restrict__ out)
{
    const int idx = blockIdx.x * blockDim.x + threadIdx.x;
    if (idx >= S_ * C_ * FR_ * C_) return;
    const int j = idx % C_;
    const int f = (idx / C_) % FR_;
    const int i = (idx / (C_ * FR_)) % C_;
    const int s = idx / (C_ * FR_ * C_);
    const int r = s ^ 1;
    const float* pa = ta + ((size_t)s * C_ + i) * D_;
    const float* pv = tv + (((size_t)r * FR_ + f) * C_ + j) * D_;
    float sa = 0.f, sv = 0.f, dot = 0.f;
    for (int d = 0; d < D_; ++d) {
        const float a = pa[d], v = pv[d];
        sa += a * a; sv += v * v; dot += a * v;
    }
    const float loss = (sa + sv - 2.0f * dot) * (1.0f / D_);
    const bool mask = (label[s * C_ + i] > 0) && (label[r * C_ + j] > 0) && (i != j);
    out[idx] = mask ? loss : 0.0f;
}

// ---------------------------------------------------------------------------
// Host launcher
// ---------------------------------------------------------------------------
extern "C" void kernel_launch(void* const* d_in, const int* in_sizes, int n_in,
                              void* d_out, int out_size, void* d_ws, size_t ws_size,
                              hipStream_t stream)
{
    const float* feat_a = (const float*)d_in[0];
    const float* feat_v = (const float*)d_in[1];
    const int*   label  = (const int*)  d_in[2];
    const float* cam    = (const float*)d_in[3];
    const float* Wt     = (const float*)d_in[4];
    const float* bt     = (const float*)d_in[5];
    const float* Ws     = (const float*)d_in[6];
    const float* bs     = (const float*)d_in[7];
    const float* Wa     = (const float*)d_in[8];
    const float* ba     = (const float*)d_in[9];
    const float* Wv     = (const float*)d_in[10];
    const float* bv     = (const float*)d_in[11];

    // Workspace layout (bytes)
    char* ws = (char*)d_ws;
    constexpr size_t OFF_WT   = 0;                                         // bf16 512x1024
    constexpr size_t OFF_WS   = OFF_WT   + (size_t)CMID_ * CIN_ * 2;
    constexpr size_t OFF_WA   = OFF_WS   + (size_t)CMID_ * CIN_ * 2;       // bf16 128x512
    constexpr size_t OFF_WV   = OFF_WA   + (size_t)D_ * CMID_ * 2;
    constexpr size_t OFF_A    = OFF_WV   + (size_t)D_ * CMID_ * 2;         // a: 896x512 f32
    constexpr size_t OFF_VT   = OFF_A    + (size_t)B_A * CMID_ * 4;        // vT: 256x512x224 bf16
    constexpr size_t OFF_CSUM = OFF_VT   + (size_t)B_V * CMID_ * NHWP_ * 2;
    constexpr size_t OFF_FG   = OFF_CSUM + (size_t)B_V * C_ * 4;           // fg: 7168x512 f32
    constexpr size_t OFF_TA   = OFF_FG   + (size_t)ROWS_TV * CMID_ * 4;    // ta: 896x128 f32
    constexpr size_t OFF_TV   = OFF_TA   + (size_t)B_A * D_ * 4;           // tv: 7168x128 f32

    unsigned short* WtBf   = (unsigned short*)(ws + OFF_WT);
    unsigned short* WsBf   = (unsigned short*)(ws + OFF_WS);
    unsigned short* WaBf   = (unsigned short*)(ws + OFF_WA);
    unsigned short* WvBf   = (unsigned short*)(ws + OFF_WV);
    float*          aOut   = (float*)(ws + OFF_A);
    unsigned short* vT     = (unsigned short*)(ws + OFF_VT);
    float*          camsum = (float*)(ws + OFF_CSUM);
    float*          fg     = (float*)(ws + OFF_FG);
    float*          ta     = (float*)(ws + OFF_TA);
    float*          tv     = (float*)(ws + OFF_TV);

    k_cvt<<<(CMID_ * CIN_) / 256, 256, 0, stream>>>(Wt, WtBf, CMID_ * CIN_);
    k_cvt<<<(CMID_ * CIN_) / 256, 256, 0, stream>>>(Ws, WsBf, CMID_ * CIN_);
    k_cvt<<<(D_ * CMID_) / 256, 256, 0, stream>>>(Wa, WaBf, D_ * CMID_);
    k_cvt<<<(D_ * CMID_) / 256, 256, 0, stream>>>(Wv, WvBf, D_ * CMID_);

    k_gemm_a<<<B_A, 256, 0, stream>>>(feat_a, WtBf, bt, aOut);
    k_gemm_v<<<dim3(B_V, NHWP_ / 16), 256, 0, stream>>>(feat_v, WsBf, bs, vT);
    k_camsum<<<B_V, 32, 0, stream>>>(cam, camsum);
    k_fg<<<B_V, 256, 0, stream>>>(cam, vT, camsum, fg);
    k_proj<<<B_A / 16, 256, 0, stream>>>(aOut, WaBf, ba, ta);
    k_proj<<<ROWS_TV / 16, 256, 0, stream>>>(fg, WvBf, bv, tv);

    float* out_co = (float*)d_out;
    float* out_di = (float*)d_out + (size_t)S_ * FR_ * C_;
    k_loss_co<<<(S_ * FR_ * C_ + 255) / 256, 256, 0, stream>>>(ta, tv, label, out_co);
    k_loss_di<<<(S_ * C_ * FR_ * C_ + 255) / 256, 256, 0, stream>>>(ta, tv, label, out_di);
}
